// MultiHeadAttention_10986526343487
// MI455X (gfx1250) — matrix operand order repaired and checked by reading, not compiled, not run on hardware
//
#include <hip/hip_runtime.h>

// Problem constants (from reference): B=2, S=2048, D=1024, H=16, dk=64.
#define B_ 2
#define S_ 2048
#define D_ 1024
#define H_ 16
#define DK_ 64
#define NEG_INF_ (-1.0e9f)

typedef __attribute__((ext_vector_type(16))) __bf16 v16bf;
typedef __attribute__((ext_vector_type(8)))  float  v8f;
typedef __attribute__((ext_vector_type(4)))  unsigned int u32x4;
typedef __attribute__((ext_vector_type(8)))  int          i32x8;
typedef __attribute__((ext_vector_type(4)))  int          i32x4;

#define GAS __attribute__((address_space(1)))
#define LAS __attribute__((address_space(3)))

#ifndef __has_builtin
#define __has_builtin(x) 0
#endif
#if __has_builtin(__builtin_amdgcn_global_load_async_to_lds_b128) && \
    __has_builtin(__builtin_amdgcn_s_wait_asynccnt)
#define HAVE_ASYNC 1
#else
#define HAVE_ASYNC 0
#endif
#if __has_builtin(__builtin_amdgcn_tensor_load_to_lds) && \
    __has_builtin(__builtin_amdgcn_s_wait_tensorcnt)
#define HAVE_TDM 1
#else
#define HAVE_TDM 0
#endif

struct alignas(16) U4  { unsigned int   x, y, z, w; };
struct alignas(8)  US4 { unsigned short s0, s1, s2, s3; };

union Frag {
  v16bf v;
  U4 q[2];
  unsigned short u[16];
};

__device__ __forceinline__ unsigned short f32_to_bf16(float f) {
  unsigned int u = __float_as_uint(f);
  u += 0x7FFFu + ((u >> 16) & 1u);   // round-to-nearest-even
  return (unsigned short)(u >> 16);
}

// 16B global -> LDS copy: async-to-LDS on gfx1250, plain copy otherwise/host.
__device__ __forceinline__ void cp16_to_lds(unsigned short* ldst,
                                            const unsigned short* gsrc) {
#if HAVE_ASYNC
  __builtin_amdgcn_global_load_async_to_lds_b128((GAS i32x4*)gsrc,
                                                 (LAS i32x4*)ldst, 0, 0);
#else
  *(U4*)ldst = *(const U4*)gsrc;
#endif
}

__device__ __forceinline__ void async_join_barrier() {
#if HAVE_ASYNC
  __builtin_amdgcn_s_wait_asynccnt(0);
#endif
  __syncthreads();
}

// ---------------------------------------------------------------------------
// Elementwise f32 -> bf16 conversion (vectorized 16B loads / 8B stores)
// ---------------------------------------------------------------------------
__global__ void cvt_f32_to_bf16(const float* __restrict__ src,
                                unsigned short* __restrict__ dst, int n4) {
  const int stride = gridDim.x * blockDim.x;
  for (int i = blockIdx.x * blockDim.x + threadIdx.x; i < n4; i += stride) {
    const float4 f = *(const float4*)(src + (size_t)i * 4);
    US4 o;
    o.s0 = f32_to_bf16(f.x); o.s1 = f32_to_bf16(f.y);
    o.s2 = f32_to_bf16(f.z); o.s3 = f32_to_bf16(f.w);
    *(US4*)(dst + (size_t)i * 4) = o;
  }
}

// ---------------------------------------------------------------------------
// Y = X @ W^T   (X: [4096,1024] bf16, W: [1024,1024] bf16)
// mode 0/1: bf16 head layout [B,H,S,dk]; mode 2: bf16 [B,H,dk,S] (V);
// mode 3: f32 merged [4096,1024] (output projection).
// 256 threads = 8 waves; block tile 128(M) x 64(N); each wave 16x64.
// W tile staged by the Tensor Data Mover (with LDS pad = 80B row stride)
// when available, else per-thread async-to-LDS, else plain copies.
// ---------------------------------------------------------------------------
__global__ __launch_bounds__(256)
void gemm_xwT(const unsigned short* __restrict__ X,
              const unsigned short* __restrict__ W,
              unsigned short* __restrict__ outB,
              float* __restrict__ outF, int mode) {
  __shared__ unsigned short wTile[64 * 40];   // 64 W-rows x 32 k, stride 40
  const int lane = threadIdx.x & 31;
  const int wave = threadIdx.x >> 5;
  const int hl   = lane >> 4;       // lane half (K split in fragment layouts)
  const int l16  = lane & 15;
  const int m0 = blockIdx.x * 128;
  const int n0 = blockIdx.y * 64;

  v8f acc[4] = {};

  // A 16x32 layout: lanes l and l+16 share row l16 (different K chunks)
  const unsigned short* Xrow = X + (size_t)(m0 + wave * 16 + l16) * D_;

  for (int k0 = 0; k0 < D_; k0 += 32) {
    __builtin_prefetch(Xrow + k0 + 32, 0, 3);   // speculative; safe past end

#if HAVE_TDM
    if (wave == 0) {
      // 2-D TDM descriptor: tile 64 rows x 32 bf16, global row stride 1024,
      // LDS padding 4 DWORDs after every 16 DWORDs -> 80B LDS row stride.
      const unsigned long long ga =
          (unsigned long long)(const void*)(W + (size_t)n0 * D_ + k0);
      u32x4 g0;
      g0[0] = 1u;                                            // count=1
      g0[1] = (unsigned)(size_t)(LAS unsigned short*)wTile;  // lds_addr
      g0[2] = (unsigned)(ga & 0xFFFFFFFFu);                  // global_addr lo
      g0[3] = (unsigned)((ga >> 32) & 0x1FFFFFFu) | (2u << 30); // hi | type=2
      i32x8 g1;
      g1[0] = (1 << 16)      // data_size = 2 bytes
            | (1 << 20)      // pad_enable
            | (3 << 22)      // pad_interval: 16 DWORDs
            | (3 << 25);     // pad_amount: 4 DWORDs
      g1[1] = (32 << 16);    // tensor_dim0 = 32
      g1[2] = (64 << 16);    // tensor_dim1 = 64
      g1[3] = (32 << 16);    // tile_dim0 = 32
      g1[4] = 64;            // tile_dim1 = 64
      g1[5] = D_;            // tensor_dim0_stride = 1024
      g1[6] = 0;
      g1[7] = 0;
      i32x4 gz = (i32x4)0;
#if defined(__clang_major__) && __clang_major__ >= 23
      __builtin_amdgcn_tensor_load_to_lds(g0, g1, gz, gz, (i32x8)0, 0);
#else
      __builtin_amdgcn_tensor_load_to_lds(g0, g1, gz, gz, 0);
#endif
      __builtin_amdgcn_s_wait_tensorcnt(0);
    }
    __syncthreads();
#else
    {
      const int row = threadIdx.x >> 2, ch = threadIdx.x & 3;
      cp16_to_lds(&wTile[row * 40 + ch * 8],
                  W + (size_t)(n0 + row) * D_ + k0 + ch * 8);
    }
    async_join_barrier();
#endif

    Frag a;                                   // A 16x32 bf16 fragment
    a.q[0] = *(const U4*)(Xrow + k0 + hl * 8);           // K = hl*8 .. +7
    a.q[1] = *(const U4*)(Xrow + k0 + 16 + hl * 8);      // K = 16+hl*8 .. +7

#pragma unroll
    for (int nt = 0; nt < 4; ++nt) {
      Frag b;                                 // B 32x16: lane=col n, K=hl*8..+7, 16+hl*8..+7
      const unsigned short* wp = &wTile[(nt * 16 + l16) * 40 + hl * 8];
      b.q[0] = *(const U4*)(wp);
      b.q[1] = *(const U4*)(wp + 16);
      acc[nt] = __builtin_amdgcn_wmma_f32_16x16x32_bf16(
          false, a.v, false, b.v, (short)0, acc[nt], false, false);
    }
    __syncthreads();
  }

  // C layout: VGPR r -> row hl*8 + r, column l16 within 16x16 tile
#pragma unroll
  for (int nt = 0; nt < 4; ++nt) {
    const int n = n0 + nt * 16 + l16;
    const int h = n >> 6, d = n & 63;
#pragma unroll
    for (int r = 0; r < 8; ++r) {
      const int m = m0 + wave * 16 + hl * 8 + r;
      const float val = acc[nt][r];
      if (mode == 3) {
        outF[(size_t)m * D_ + n] = val;
      } else {
        const int bb = m >> 11, s = m & (S_ - 1);
        size_t idx;
        if (mode == 2) idx = ((size_t)(bb * H_ + h) * DK_ + d) * S_ + s;   // Vt
        else           idx = ((size_t)(bb * H_ + h) * S_ + s) * DK_ + d;   // Qh/Kh
        outB[idx] = f32_to_bf16(val);
      }
    }
  }
}

// ---------------------------------------------------------------------------
// Flash attention: per (b,h) and 64-query tile; 4 waves x 16 query rows.
// scores = (Q K^T) * sqrt(dk) [ref divides by dk^-0.5], key mask -> -1e9,
// online softmax, O += P V, O /= l.  Qh/Kh: [B,H,S,dk], Vt: [B,H,dk,S].
// K/V blocks staged with async-to-LDS (ASYNCcnt) when available.
// ---------------------------------------------------------------------------
__global__ __launch_bounds__(128)
void flash_attn(const unsigned short* __restrict__ Qh,
                const unsigned short* __restrict__ Kh,
                const unsigned short* __restrict__ Vt,
                const unsigned char* __restrict__ mask,
                unsigned short* __restrict__ ctx) {
  __shared__ unsigned short kT[64 * 72];   // 64 keys x 64 d
  __shared__ unsigned short vT[64 * 72];   // 64 d    x 64 keys
  __shared__ unsigned short pT[64 * 72];   // 4 waves x 16 rows x 64 keys
  const int lane = threadIdx.x & 31, wave = threadIdx.x >> 5;
  const int hl = lane >> 4, l16 = lane & 15;
  const int bh = blockIdx.y, bb = bh >> 4, h = bh & (H_ - 1);
  const int q0 = blockIdx.x * 64;

  // Q fragments (dk=64 -> two 16x32 A-frags), kept in registers for all blocks
  const unsigned short* Qrow = Qh + ((size_t)bh * S_ + q0 + wave * 16 + l16) * DK_;
  Frag qa[2];
#pragma unroll
  for (int t = 0; t < 2; ++t) {
    qa[t].q[0] = *(const U4*)(Qrow + t * 32 + hl * 8);
    qa[t].q[1] = *(const U4*)(Qrow + t * 32 + 16 + hl * 8);
  }

  float mrow[8], lrow[8];
  v8f o[4] = {};
#pragma unroll
  for (int r = 0; r < 8; ++r) { mrow[r] = NEG_INF_; lrow[r] = 0.f; }

  for (int kb0 = 0; kb0 < S_; kb0 += 64) {
    // Stage K block (keys x d) and V block (d x keys) into LDS
    for (int c = threadIdx.x; c < 512; c += 128) {
      const int row = c >> 3, ch = c & 7;
      cp16_to_lds(&kT[row * 72 + ch * 8],
                  Kh + ((size_t)bh * S_ + kb0 + row) * DK_ + ch * 8);
      cp16_to_lds(&vT[row * 72 + ch * 8],
                  Vt + ((size_t)bh * DK_ + row) * S_ + kb0 + ch * 8);
    }
    __builtin_prefetch(Kh + ((size_t)bh * S_ + kb0 + 64) * DK_, 0, 3);
    async_join_barrier();

    // scores: 16 (q rows) x 64 (keys) = 4 N-tiles, dk=64 -> 2 K-steps
    v8f sc[4] = {};
#pragma unroll
    for (int t = 0; t < 2; ++t) {
#pragma unroll
      for (int nt = 0; nt < 4; ++nt) {
        Frag kb_;   // B 32x16: lane col = key, K(d) = t*32 + hl*8 .. +7, t*32+16+hl*8 .. +7
        const unsigned short* kp = &kT[(nt * 16 + l16) * 72 + t * 32 + hl * 8];
        kb_.q[0] = *(const U4*)kp;
        kb_.q[1] = *(const U4*)(kp + 16);
        sc[nt] = __builtin_amdgcn_wmma_f32_16x16x32_bf16(
            false, qa[t].v, false, kb_.v, (short)0, sc[nt], false, false);
      }
    }

    // scale (ref multiplies scores by sqrt(dk)=8) + key mask; lane = one key col
#pragma unroll
    for (int nt = 0; nt < 4; ++nt) {
      const bool mk = mask[(size_t)bb * S_ + kb0 + nt * 16 + l16] != 0;
#pragma unroll
      for (int r = 0; r < 8; ++r)
        sc[nt][r] = mk ? NEG_INF_ : sc[nt][r] * 8.0f;
    }

    // online softmax: row (hl*8+r) lives entirely in this lane's 16-lane half
#pragma unroll
    for (int r = 0; r < 8; ++r) {
      float v = fmaxf(fmaxf(sc[0][r], sc[1][r]), fmaxf(sc[2][r], sc[3][r]));
#pragma unroll
      for (int off = 1; off < 16; off <<= 1)
        v = fmaxf(v, __shfl_xor(v, off, 32));
      const float mnew = fmaxf(mrow[r], v);
      const float fac = __expf(mrow[r] - mnew);
      float rs = 0.f;
#pragma unroll
      for (int nt = 0; nt < 4; ++nt) {
        const float p = __expf(sc[nt][r] - mnew);
        sc[nt][r] = p;
        rs += p;
        o[nt][r] *= fac;
      }
#pragma unroll
      for (int off = 1; off < 16; off <<= 1)
        rs += __shfl_xor(rs, off, 32);
      lrow[r] = lrow[r] * fac + rs;
      mrow[r] = mnew;
    }

    // C-layout P -> LDS -> A-layout fragments (per-wave private region;
    // same-wave LDS ops complete in order, no barrier needed)
    unsigned short* pW = &pT[wave * 16 * 72];
#pragma unroll
    for (int nt = 0; nt < 4; ++nt)
#pragma unroll
      for (int r = 0; r < 8; ++r)
        pW[(hl * 8 + r) * 72 + nt * 16 + l16] = f32_to_bf16(sc[nt][r]);

    const unsigned short* pR = &pT[(wave * 16 + l16) * 72];
#pragma unroll
    for (int t = 0; t < 2; ++t) {
      Frag pa;
      pa.q[0] = *(const U4*)(pR + t * 32 + hl * 8);
      pa.q[1] = *(const U4*)(pR + t * 32 + 16 + hl * 8);
#pragma unroll
      for (int nt = 0; nt < 4; ++nt) {
        Frag vb_;  // B 32x16: lane col = d, K(keys) = t*32 + hl*8 .. +7, t*32+16+hl*8 .. +7
        const unsigned short* vp = &vT[(nt * 16 + l16) * 72 + t * 32 + hl * 8];
        vb_.q[0] = *(const U4*)vp;
        vb_.q[1] = *(const U4*)(vp + 16);
        o[nt] = __builtin_amdgcn_wmma_f32_16x16x32_bf16(
            false, pa.v, false, vb_.v, (short)0, o[nt], false, false);
      }
    }
    __syncthreads();
  }

  // normalize and store ctx (bf16, merged heads layout [B,S,D])
#pragma unroll
  for (int nt = 0; nt < 4; ++nt) {
#pragma unroll
    for (int r = 0; r < 8; ++r) {
      const int srow = q0 + wave * 16 + hl * 8 + r;
      const int col  = h * 64 + nt * 16 + l16;
      const float val = o[nt][r] / lrow[r];
      ctx[((size_t)(bb * S_ + srow)) * D_ + col] = f32_to_bf16(val);
    }
  }
}

// ---------------------------------------------------------------------------
// Host launch
// ---------------------------------------------------------------------------
extern "C" void kernel_launch(void* const* d_in, const int* in_sizes, int n_in,
                              void* d_out, int out_size, void* d_ws, size_t ws_size,
                              hipStream_t stream) {
  (void)in_sizes; (void)n_in; (void)out_size; (void)ws_size;
  const float* q  = (const float*)d_in[0];
  const float* k  = (const float*)d_in[1];
  const float* v  = (const float*)d_in[2];
  const unsigned char* mask = (const unsigned char*)d_in[3];
  const float* Wq = (const float*)d_in[4];
  const float* Wk = (const float*)d_in[5];
  const float* Wv = (const float*)d_in[6];
  const float* Wo = (const float*)d_in[7];
  float* out = (float*)d_out;

  // Workspace (bf16): 3*8MB qkv + 4*2MB weights + 3*8MB heads + 8MB ctx = 64MB
  unsigned short* ws = (unsigned short*)d_ws;
  const size_t NX = (size_t)B_ * S_ * D_;   // 4M elems
  const size_t NW = (size_t)D_ * D_;        // 1M elems
  unsigned short* qb  = ws;
  unsigned short* kb  = qb + NX;
  unsigned short* vb  = kb + NX;
  unsigned short* wqb = vb + NX;
  unsigned short* wkb = wqb + NW;
  unsigned short* wvb = wkb + NW;
  unsigned short* wob = wvb + NW;
  unsigned short* Qh  = wob + NW;
  unsigned short* Kh  = Qh + NX;
  unsigned short* Vt  = Kh + NX;
  unsigned short* ctx = Vt + NX;

  cvt_f32_to_bf16<<<dim3(1024), dim3(256), 0, stream>>>(q,  qb,  (int)(NX / 4));
  cvt_f32_to_bf16<<<dim3(1024), dim3(256), 0, stream>>>(k,  kb,  (int)(NX / 4));
  cvt_f32_to_bf16<<<dim3(1024), dim3(256), 0, stream>>>(v,  vb,  (int)(NX / 4));
  cvt_f32_to_bf16<<<dim3(256),  dim3(256), 0, stream>>>(Wq, wqb, (int)(NW / 4));
  cvt_f32_to_bf16<<<dim3(256),  dim3(256), 0, stream>>>(Wk, wkb, (int)(NW / 4));
  cvt_f32_to_bf16<<<dim3(256),  dim3(256), 0, stream>>>(Wv, wvb, (int)(NW / 4));
  cvt_f32_to_bf16<<<dim3(256),  dim3(256), 0, stream>>>(Wo, wob, (int)(NW / 4));

  const dim3 gg(32, 16), gb(256);
  gemm_xwT<<<gg, gb, 0, stream>>>(qb, wqb, Qh, nullptr, 0);
  gemm_xwT<<<gg, gb, 0, stream>>>(kb, wkb, Kh, nullptr, 1);
  gemm_xwT<<<gg, gb, 0, stream>>>(vb, wvb, Vt, nullptr, 2);

  flash_attn<<<dim3(S_ / 64, B_ * H_), dim3(128), 0, stream>>>(Qh, Kh, Vt, mask, ctx);

  gemm_xwT<<<gg, gb, 0, stream>>>(ctx, wob, nullptr, out, 3);
}
